// SimpleMoELayer_90572270338151
// MI455X (gfx1250) — matrix operand order repaired and checked by reading, not compiled, not run on hardware
//
#include <hip/hip_runtime.h>

typedef __attribute__((ext_vector_type(16))) _Float16 v16h;
typedef __attribute__((ext_vector_type(8)))  float    v8f;

#define NUM_EXPERTS 8
#define D_MODEL     768
#define D_FF        3072
#define TOKENS      2048   // 2 * 1024

// -------------------- small helper kernels --------------------

__global__ void zero_kernel(int* __restrict__ cnt) {
    if (threadIdx.x < NUM_EXPERTS) cnt[threadIdx.x] = 0;
}

__global__ __launch_bounds__(256) void cvt_kernel(const float* __restrict__ x,
                                                  _Float16* __restrict__ xh) {
    int i = blockIdx.x * 256 + threadIdx.x;   // grid sized exactly to TOKENS*D_MODEL
    xh[i] = (_Float16)x[i];
}

// Per-token router: logits = x @ rw + rb ; top-1 argmax (softmax is monotone,
// output is not prob-scaled in the reference). First-max wins like jnp.argmax.
__global__ __launch_bounds__(256) void router_kernel(const float* __restrict__ x,
                                                     const float* __restrict__ rw,
                                                     const float* __restrict__ rb,
                                                     int* __restrict__ cnt,
                                                     int* __restrict__ bucket) {
    int t = blockIdx.x * blockDim.x + threadIdx.x;
    if (t >= TOKENS) return;
    float acc[NUM_EXPERTS];
#pragma unroll
    for (int e = 0; e < NUM_EXPERTS; ++e) acc[e] = rb[e];
    const float* xr = x + (size_t)t * D_MODEL;
    for (int k = 0; k < D_MODEL; ++k) {
        float xv = xr[k];
#pragma unroll
        for (int e = 0; e < NUM_EXPERTS; ++e) acc[e] += xv * rw[k * NUM_EXPERTS + e];
    }
    int best = 0; float bv = acc[0];
#pragma unroll
    for (int e = 1; e < NUM_EXPERTS; ++e) {
        if (acc[e] > bv) { bv = acc[e]; best = e; }
    }
    int pos = atomicAdd(&cnt[best], 1);
    bucket[best * TOKENS + pos] = t;
}

__global__ void loss_kernel(const int* __restrict__ cnt, float* __restrict__ out) {
    if (threadIdx.x == 0) {
        float s = 0.f;
#pragma unroll
        for (int e = 0; e < NUM_EXPERTS; ++e) {
            float u = (float)cnt[e] * (1.0f / (float)TOKENS) - 0.125f;
            s += u * u;
        }
        out[0] = s * (1.0f / (float)NUM_EXPERTS);
    }
}

// -------------------- WMMA expert GEMM --------------------
// C[tok, n] = act( gatherA[tok, :Kdim] @ W[e, :Kdim, :Ndim] + bias[e, :Ndim] )
// 128 threads = 4 wave32 in 2x2; block tile 32(M) x 64(N), K-step 32.
// A staged row-major [m][k]; B staged TRANSPOSED [n][k] so that both A and B
// wave fragments are contiguous 128-bit LDS loads per the CDNA5 WMMA layouts.
// B staging does a 4x4 register transpose so the LDS scatter is b64 stores.

#define TM 32
#define TN 64
#define TK 32
#define LSTR 48   // LDS row stride in halfs: 96B (16B-aligned, non-pow2 banks)

union V16 { v16h v; uint4 q[2]; };

template <int Kdim, int Ndim, bool RELU, typename OutT>
__global__ __launch_bounds__(128) void moe_gemm(const _Float16* __restrict__ Ain,  // [TOKENS, Kdim]
                                                const float* __restrict__ W,       // [E, Kdim, Ndim]
                                                const float* __restrict__ bias,    // [E, Ndim]
                                                const int* __restrict__ cnt,
                                                const int* __restrict__ bucket,
                                                OutT* __restrict__ Out) {          // [TOKENS, Ndim]
    const int e  = blockIdx.z;
    const int n0 = blockIdx.x * TN;
    const int m0 = blockIdx.y * TM;
    const int tcount = cnt[e];
    if (m0 >= tcount) return;     // uniform per block -> EXEC all-ones for WMMA

    __shared__ _Float16 As[TM][LSTR];   // [m][k]
    __shared__ _Float16 Bs[TN][LSTR];   // [n][k]  (transposed weight tile)
    __shared__ int toks[TM];

    const int tid  = threadIdx.x;
    const int lane = tid & 31;
    const int wave = tid >> 5;
    const int wm = wave >> 1;       // 0..1  (16-row slab)
    const int wn = wave & 1;        // 0..1  (32-col slab)
    const int lo = lane & 15;
    const int hi = lane >> 4;

    if (tid < TM) {
        int idx = m0 + tid;
        if (idx >= tcount) idx = tcount - 1;            // clamp; stores are guarded
        toks[tid] = bucket[e * TOKENS + idx];
    }
    __syncthreads();

    // ---- staging thread mappings (constant across K loop) ----
    // A: 1 row x 8 halfs per thread
    const int ar = tid >> 2;            // 0..31 : A row (m)
    const int ac = (tid & 3) * 8;       // 0,8,16,24 : A k-offset
    // B: 4 k-rows x 4 n per thread (register 4x4 transpose -> b64 LDS stores)
    const int kq = tid >> 4;            // 0..7  : k-quad (rows 4kq..4kq+3)
    const int nc = (tid & 15) * 4;      // 0..60 : n-offset
    const _Float16* aSrcBase = Ain + (size_t)toks[ar] * Kdim + ac;
    const float*    wSrcBase = W + (size_t)e * Kdim * Ndim
                                 + (size_t)(4 * kq) * Ndim + n0 + nc;

    v8f acc0 = {};
    v8f acc1 = {};

    // ---- software pipeline: prefetch K-step 0 ----
    uint4  aReg = *(const uint4*)(aSrcBase);                 // 8 halfs
    float4 bReg[4];                                          // 4 k-rows x 4 n
#pragma unroll
    for (int r = 0; r < 4; ++r)
        bReg[r] = *(const float4*)(wSrcBase + (size_t)r * Ndim);

    for (int k0 = 0; k0 < Kdim; k0 += TK) {
        // ---- commit staged registers to LDS ----
        *(uint4*)&As[ar][ac] = aReg;                         // b128 store
        {
            _Float16 hp[4][4];                               // [n][k] in registers
#pragma unroll
            for (int r = 0; r < 4; ++r) {
                hp[0][r] = (_Float16)bReg[r].x;
                hp[1][r] = (_Float16)bReg[r].y;
                hp[2][r] = (_Float16)bReg[r].z;
                hp[3][r] = (_Float16)bReg[r].w;
            }
#pragma unroll
            for (int n = 0; n < 4; ++n)                      // b64 stores, k-contiguous
                *(uint2*)&Bs[nc + n][4 * kq] = *(uint2*)&hp[n][0];
        }
        __syncthreads();

        // ---- prefetch next K-step while LDS is consumed ----
        if (k0 + TK < Kdim) {
            aReg = *(const uint4*)(aSrcBase + k0 + TK);
            const float* ws = wSrcBase + (size_t)(k0 + TK) * Ndim;
#pragma unroll
            for (int r = 0; r < 4; ++r)
                bReg[r] = *(const float4*)(ws + (size_t)r * Ndim);
        }

        // ---- wave fragments: all contiguous 128-bit LDS loads ----
        V16 ua, ub0, ub1;
        const _Float16* arow  = &As[(wm << 4) + lo][0];
        const _Float16* b0row = &Bs[(wn << 5) + lo][0];
        const _Float16* b1row = &Bs[(wn << 5) + 16 + lo][0];
        // A 16x32: lane<16 -> K=0..15, lane>=16 -> K=16..31
        ua.q[0]  = *(const uint4*)(arow + (hi << 4));
        ua.q[1]  = *(const uint4*)(arow + (hi << 4) + 8);
        // B 32x16: col = lane%16; lane<16 -> K=0..15, lane>=16 -> K=16..31
        ub0.q[0] = *(const uint4*)(b0row + (hi << 4));
        ub0.q[1] = *(const uint4*)(b0row + (hi << 4) + 8);
        ub1.q[0] = *(const uint4*)(b1row + (hi << 4));
        ub1.q[1] = *(const uint4*)(b1row + (hi << 4) + 8);

        acc0 = __builtin_amdgcn_wmma_f32_16x16x32_f16(false, ua.v, false, ub0.v,
                                                      (short)0, acc0, false, false);
        acc1 = __builtin_amdgcn_wmma_f32_16x16x32_f16(false, ua.v, false, ub1.v,
                                                      (short)0, acc1, false, false);
        __syncthreads();
    }

    // ---- epilogue: C/D layout -> row M = r + (lane/16)*8, col N = lane%16 ----
    const int col = n0 + (wn << 5) + lo;
    const float b0 = bias[e * Ndim + col];
    const float b1v = bias[e * Ndim + col + 16];
    if (m0 + TM <= tcount) {
        // fast path: full tile, no per-row guards (uniform branch)
#pragma unroll
        for (int r = 0; r < 8; ++r) {
            int mrow = (wm << 4) + (hi << 3) + r;
            int tok = toks[mrow];
            float v0 = acc0[r] + b0;
            float v1 = acc1[r] + b1v;
            if (RELU) { v0 = fmaxf(v0, 0.f); v1 = fmaxf(v1, 0.f); }
            Out[(size_t)tok * Ndim + col]      = (OutT)v0;
            Out[(size_t)tok * Ndim + col + 16] = (OutT)v1;
        }
    } else {
#pragma unroll
        for (int r = 0; r < 8; ++r) {
            int mrow = (wm << 4) + (hi << 3) + r;
            if (m0 + mrow < tcount) {
                int tok = toks[mrow];
                float v0 = acc0[r] + b0;
                float v1 = acc1[r] + b1v;
                if (RELU) { v0 = fmaxf(v0, 0.f); v1 = fmaxf(v1, 0.f); }
                Out[(size_t)tok * Ndim + col]      = (OutT)v0;
                Out[(size_t)tok * Ndim + col + 16] = (OutT)v1;
            }
        }
    }
}

// -------------------- launch --------------------

extern "C" void kernel_launch(void* const* d_in, const int* in_sizes, int n_in,
                              void* d_out, int out_size, void* d_ws, size_t ws_size,
                              hipStream_t stream) {
    const float* x  = (const float*)d_in[0];
    const float* rw = (const float*)d_in[1];
    const float* rb = (const float*)d_in[2];
    const float* w1 = (const float*)d_in[3];
    const float* b1 = (const float*)d_in[4];
    const float* w2 = (const float*)d_in[5];
    const float* b2 = (const float*)d_in[6];
    float* out = (float*)d_out;

    char* ws = (char*)d_ws;
    int*      cnt    = (int*)ws;                                   // 8 ints
    int*      bucket = (int*)(ws + 128);                           // 8*2048 ints = 64 KB
    _Float16* xh     = (_Float16*)(ws + 128 + 65536);              // 2048*768  f16 = 3 MB
    _Float16* H      = (_Float16*)(ws + 128 + 65536
                                   + (size_t)TOKENS * D_MODEL * 2); // 2048*3072 f16 = 12 MB

    zero_kernel<<<1, 32, 0, stream>>>(cnt);
    cvt_kernel<<<(TOKENS * D_MODEL) / 256, 256, 0, stream>>>(x, xh);
    router_kernel<<<TOKENS / 256, 256, 0, stream>>>(x, rw, rb, cnt, bucket);
    loss_kernel<<<1, 32, 0, stream>>>(cnt, out + (size_t)TOKENS * D_MODEL);

    dim3 g1(D_FF / TN, TOKENS / TM, NUM_EXPERTS);       // (48, 64, 8)
    moe_gemm<D_MODEL, D_FF, true, _Float16>
        <<<g1, 128, 0, stream>>>(xh, w1, b1, cnt, bucket, H);

    dim3 g2(D_MODEL / TN, TOKENS / TM, NUM_EXPERTS);    // (12, 64, 8)
    moe_gemm<D_FF, D_MODEL, false, float>
        <<<g2, 128, 0, stream>>>(H, w2, b2, cnt, bucket, out);
}
